// ITASelfAttention_75935021793297
// MI455X (gfx1250) — matrix operand order repaired and checked by reading, not compiled, not run on hardware
//
#include <hip/hip_runtime.h>

typedef __bf16 bf16_t;
typedef __attribute__((ext_vector_type(16))) __bf16 v16bf;
typedef __attribute__((ext_vector_type(8)))  float  v8f;
typedef __attribute__((ext_vector_type(4)))  int    v4i_t;

#define B_  8
#define N_  1024
#define E_  768
#define P_  768
#define NH_ 12
#define HD_ 64
#define M_  (B_ * N_)   // 8192 total rows

#if defined(__gfx1250__) && __has_builtin(__builtin_amdgcn_global_load_async_to_lds_b128) && \
    __has_builtin(__builtin_amdgcn_s_wait_asynccnt)
#define HAVE_ASYNC_LDS 1
#else
#define HAVE_ASYNC_LDS 0
#endif

#if HAVE_ASYNC_LDS
typedef __attribute__((address_space(1))) v4i_t as1_v4i;  // global int4
typedef __attribute__((address_space(3))) v4i_t as3_v4i;  // LDS int4
#endif

// copy 8 bf16 (16 bytes) global -> LDS, async when available
static __device__ __forceinline__ void copy_b128_to_lds(const bf16_t* g, bf16_t* l) {
#if HAVE_ASYNC_LDS
  __builtin_amdgcn_global_load_async_to_lds_b128(
      (as1_v4i*)(size_t)(const void*)g, (as3_v4i*)(unsigned)(size_t)(void*)l, 0, 0);
#else
  *(uint4*)l = *(const uint4*)g;
#endif
}
static __device__ __forceinline__ void async_wait_all() {
#if HAVE_ASYNC_LDS
  __builtin_amdgcn_s_wait_asynccnt(0);
#endif
}

static __device__ __forceinline__ v8f wmma_bf16(v16bf a, v16bf b, v8f c) {
  // D = A(16x32 bf16) * B(32x16 bf16) + C(16x16 f32)
  return __builtin_amdgcn_wmma_f32_16x16x32_bf16(false, a, false, b, (short)0, c,
                                                 false, false);
}

// ---------------------------------------------------------------------------
// fp32 -> bf16 convert (x activations), 4 elements / thread
// ---------------------------------------------------------------------------
__global__ __launch_bounds__(256) void cvt_bf16_kernel(const float* __restrict__ in,
                                                       bf16_t* __restrict__ out, int n) {
  int i = (blockIdx.x * 256 + threadIdx.x) * 4;
  if (i + 3 < n) {
    float4 f = *(const float4*)(in + i);
    out[i + 0] = (bf16_t)f.x;
    out[i + 1] = (bf16_t)f.y;
    out[i + 2] = (bf16_t)f.z;
    out[i + 3] = (bf16_t)f.w;
  }
}

// ---------------------------------------------------------------------------
// fp32 [K,N] -> bf16 [N,K] transposed convert (weights), LDS tiled
// ---------------------------------------------------------------------------
__global__ __launch_bounds__(256) void transpose_cvt_kernel(const float* __restrict__ W,
                                                            bf16_t* __restrict__ Wt,
                                                            int K, int N) {
  __shared__ float tile[32][33];
  int bx = blockIdx.x * 32;  // N tile
  int by = blockIdx.y * 32;  // K tile
  int tx = threadIdx.x & 31;
  int ty = threadIdx.x >> 5;  // 0..7
#pragma unroll
  for (int i = 0; i < 32; i += 8)
    tile[ty + i][tx] = W[(size_t)(by + ty + i) * N + bx + tx];
  __syncthreads();
#pragma unroll
  for (int i = 0; i < 32; i += 8)
    Wt[(size_t)(bx + ty + i) * K + by + tx] = (bf16_t)tile[tx][ty + i];
}

// ---------------------------------------------------------------------------
// bf16 GEMM: C[M,N] = A[M,K] * Bt[N,K]^T + bias.  128x128 block tile, BK=32,
// double-buffered LDS with async global->LDS staging, 8 waves (4x2),
// each wave 2x4 tiles of v_wmma_f32_16x16x32_bf16.
// ---------------------------------------------------------------------------
template <bool OUT_F32>
__global__ __launch_bounds__(256) void gemm_bf16_kernel(const bf16_t* __restrict__ A,
                                                        const bf16_t* __restrict__ Bt,
                                                        const float* __restrict__ bias,
                                                        void* __restrict__ Cout,
                                                        int M, int Nn, int Kk) {
  const int bm = blockIdx.y * 128;
  const int bn = blockIdx.x * 128;
  const int tid = threadIdx.x;
  const int wid = tid >> 5;
  const int lane = tid & 31;
  const int wm = (wid & 3) * 32;   // wave row offset within block tile
  const int wn = (wid >> 2) * 64;  // wave col offset within block tile

  __shared__ __align__(16) bf16_t Asl[2][128 * 32];
  __shared__ __align__(16) bf16_t Bsl[2][128 * 32];

  v8f acc[2][4];
#pragma unroll
  for (int i = 0; i < 2; ++i)
#pragma unroll
    for (int j = 0; j < 4; ++j) acc[i][j] = (v8f){0.f, 0.f, 0.f, 0.f, 0.f, 0.f, 0.f, 0.f};

  const int frow = lane & 15;        // fragment row / col
  const int half = lane >> 4;        // lane half select
  const int kg = half * 8;           // unused (A-frag K base is kb)
  const int kb = half * 16;          // B-frag K base

  const int lr = tid >> 1;           // 0..127 staging row
  const int lc = (tid & 1) * 16;     // staging col (bf16 elems)

  auto stage = [&](int buf, int k0) {
    const bf16_t* ag = A + (size_t)(bm + lr) * Kk + k0 + lc;
    const bf16_t* bg = Bt + (size_t)(bn + lr) * Kk + k0 + lc;
    bf16_t* al = &Asl[buf][lr * 32 + lc];
    bf16_t* bl = &Bsl[buf][lr * 32 + lc];
    copy_b128_to_lds(ag, al);
    copy_b128_to_lds(ag + 8, al + 8);
    copy_b128_to_lds(bg, bl);
    copy_b128_to_lds(bg + 8, bl + 8);
  };

  const int ksteps = Kk / 32;
  stage(0, 0);
  async_wait_all();
  __syncthreads();

  for (int kk = 0; kk < ksteps; ++kk) {
    const int cur = kk & 1;
    if (kk + 1 < ksteps) stage(cur ^ 1, (kk + 1) * 32);  // overlap next panel fill

    v16bf af[2], bfr[4];
#pragma unroll
    for (int i = 0; i < 2; ++i) {
      const bf16_t* s = &Asl[cur][(wm + i * 16 + frow) * 32];
#pragma unroll
      for (int e = 0; e < 8; ++e) {
        af[i][e] = s[kb + e];
        af[i][e + 8] = s[kb + 8 + e];
      }
    }
#pragma unroll
    for (int j = 0; j < 4; ++j) {
      const bf16_t* s = &Bsl[cur][(wn + j * 16 + frow) * 32 + kb];
#pragma unroll
      for (int e = 0; e < 16; ++e) bfr[j][e] = s[e];
    }
#pragma unroll
    for (int i = 0; i < 2; ++i)
#pragma unroll
      for (int j = 0; j < 4; ++j) acc[i][j] = wmma_bf16(af[i], bfr[j], acc[i][j]);

    async_wait_all();
    __syncthreads();
  }

  // epilogue: C layout row = r + 8*half, col = lane&15
#pragma unroll
  for (int i = 0; i < 2; ++i)
#pragma unroll
    for (int j = 0; j < 4; ++j)
#pragma unroll
      for (int r = 0; r < 8; ++r) {
        int gm = bm + wm + i * 16 + r + 8 * half;
        int gn = bn + wn + j * 16 + frow;
        float v = acc[i][j][r] + bias[gn];
        if (OUT_F32)
          ((float*)Cout)[(size_t)gm * Nn + gn] = v;
        else
          ((bf16_t*)Cout)[(size_t)gm * Nn + gn] = (bf16_t)v;
      }
}

// ---------------------------------------------------------------------------
// Fused flash attention: grid (qtile=8, head=12, batch=8), 256 threads.
// Each wave owns 16 q rows; streams 128-key K/V tiles through double-buffered
// LDS (async fill) with online softmax; ctx accumulated in fp32 via WMMA.
// ---------------------------------------------------------------------------
__global__ __launch_bounds__(256) void attn_kernel(const bf16_t* __restrict__ Q,
                                                   const bf16_t* __restrict__ K,
                                                   const bf16_t* __restrict__ V,
                                                   bf16_t* __restrict__ Ctx) {
  const int qt = blockIdx.x;           // q tile (128 rows)
  const int h = blockIdx.y;            // head
  const int b = blockIdx.z;            // batch
  const int rowbase = b * N_;          // batch row offset into [8192, 768]
  const int coff = h * HD_;            // head column offset

  const int tid = threadIdx.x;
  const int wid = tid >> 5;
  const int lane = tid & 31;
  const int qrow0 = qt * 128 + wid * 16;

  __shared__ __align__(16) bf16_t Kt[2][128 * HD_];  // 32 KB
  __shared__ __align__(16) bf16_t Vt[2][128 * HD_];  // 32 KB
  __shared__ __align__(16) bf16_t Pl[8][16 * 128];   // 32 KB, per-wave P

  const int frow = lane & 15;
  const int half = lane >> 4;
  const int kg = half * 8;
  const int kb = half * 16;

  // Q fragments: 16 rows x 64 dims = 2 chunks of 16x32
  v16bf qf[2];
#pragma unroll
  for (int c = 0; c < 2; ++c) {
    const bf16_t* s = Q + (size_t)(rowbase + qrow0 + frow) * P_ + coff + c * 32;
#pragma unroll
    for (int e = 0; e < 8; ++e) {
      qf[c][e] = s[kb + e];
      qf[c][e + 8] = s[kb + 8 + e];
    }
  }

  v8f ctx[4];
#pragma unroll
  for (int ct = 0; ct < 4; ++ct) ctx[ct] = (v8f){0.f, 0.f, 0.f, 0.f, 0.f, 0.f, 0.f, 0.f};
  float ms[8], ls[8];
#pragma unroll
  for (int r = 0; r < 8; ++r) {
    ms[r] = -1e30f;
    ls[r] = 0.f;
  }

  const int lr = tid >> 1;          // 0..127 staging row
  const int lc = (tid & 1) * 32;    // staging col (bf16 elems)

  auto stage = [&](int buf, int j) {
    const bf16_t* kgp = K + (size_t)(rowbase + j + lr) * P_ + coff + lc;
    const bf16_t* vgp = V + (size_t)(rowbase + j + lr) * P_ + coff + lc;
    bf16_t* kl = &Kt[buf][lr * HD_ + lc];
    bf16_t* vl = &Vt[buf][lr * HD_ + lc];
#pragma unroll
    for (int q = 0; q < 4; ++q) {
      copy_b128_to_lds(kgp + q * 8, kl + q * 8);
      copy_b128_to_lds(vgp + q * 8, vl + q * 8);
    }
  };

  stage(0, 0);
  async_wait_all();
  __syncthreads();

  for (int jt = 0; jt < N_ / 128; ++jt) {
    const int cur = jt & 1;
    if (jt + 1 < N_ / 128) stage(cur ^ 1, (jt + 1) * 128);  // overlap next K/V fill

    // S = Q(16x64) @ K^T(64x128): 8 key sub-tiles, 2 WMMA each
    v8f sc[8];
#pragma unroll
    for (int kt = 0; kt < 8; ++kt) {
      sc[kt] = (v8f){0.f, 0.f, 0.f, 0.f, 0.f, 0.f, 0.f, 0.f};
#pragma unroll
      for (int c = 0; c < 2; ++c) {
        v16bf bfr;
        const bf16_t* s = &Kt[cur][(kt * 16 + frow) * HD_ + c * 32 + kb];
#pragma unroll
        for (int e = 0; e < 16; ++e) bfr[e] = s[e];
        sc[kt] = wmma_bf16(qf[c], bfr, sc[kt]);
      }
    }

    // online softmax (rows live in one 16-lane half; reduce with shfl_xor)
    float al[8];
#pragma unroll
    for (int r = 0; r < 8; ++r) {
      float tm = -1e30f;
#pragma unroll
      for (int kt = 0; kt < 8; ++kt) tm = fmaxf(tm, sc[kt][r]);
#pragma unroll
      for (int off = 1; off < 16; off <<= 1) tm = fmaxf(tm, __shfl_xor(tm, off, 32));
      float mn = fmaxf(ms[r], tm);
      al[r] = __expf(ms[r] - mn);
      ms[r] = mn;
      float rs = 0.f;
#pragma unroll
      for (int kt = 0; kt < 8; ++kt) {
        float pv = __expf(sc[kt][r] - mn);
        sc[kt][r] = pv;
        rs += pv;
      }
#pragma unroll
      for (int off = 1; off < 16; off <<= 1) rs += __shfl_xor(rs, off, 32);
      ls[r] = ls[r] * al[r] + rs;
    }
#pragma unroll
    for (int ct = 0; ct < 4; ++ct)
#pragma unroll
      for (int r = 0; r < 8; ++r) ctx[ct][r] *= al[r];

    // P (C-layout f32) -> per-wave LDS bf16 [16][128], then reload as A-frags
    bf16_t* pw = &Pl[wid][0];
#pragma unroll
    for (int kt = 0; kt < 8; ++kt)
#pragma unroll
      for (int r = 0; r < 8; ++r)
        pw[(r + 8 * half) * 128 + kt * 16 + frow] = (bf16_t)sc[kt][r];

    // ctx += P(16x128) @ V(128x64): 4 K-chunks x 4 d-tiles
#pragma unroll
    for (int kc = 0; kc < 4; ++kc) {
      v16bf pf;
      const bf16_t* ps = pw + frow * 128 + kc * 32;
#pragma unroll
      for (int e = 0; e < 8; ++e) {
        pf[e] = ps[kb + e];
        pf[e + 8] = ps[kb + 8 + e];
      }
#pragma unroll
      for (int ct = 0; ct < 4; ++ct) {
        v16bf vf;
#pragma unroll
        for (int e = 0; e < 16; ++e)
          vf[e] = Vt[cur][(kc * 32 + kb + e) * HD_ + ct * 16 + frow];
        ctx[ct] = wmma_bf16(pf, vf, ctx[ct]);
      }
    }

    async_wait_all();
    __syncthreads();
  }

  // normalize and store ctx (bf16) into [8192, 768] at head columns
  float inv[8];
#pragma unroll
  for (int r = 0; r < 8; ++r) inv[r] = 1.f / ls[r];
#pragma unroll
  for (int ct = 0; ct < 4; ++ct)
#pragma unroll
    for (int r = 0; r < 8; ++r)
      Ctx[(size_t)(rowbase + qrow0 + r + 8 * half) * P_ + coff + ct * 16 + frow] =
          (bf16_t)(ctx[ct][r] * inv[r]);
}

// ---------------------------------------------------------------------------
extern "C" void kernel_launch(void* const* d_in, const int* in_sizes, int n_in,
                              void* d_out, int out_size, void* d_ws, size_t ws_size,
                              hipStream_t stream) {
  (void)in_sizes; (void)n_in; (void)out_size; (void)ws_size;
  const float* x  = (const float*)d_in[0];
  const float* wq = (const float*)d_in[1];
  const float* bq = (const float*)d_in[2];
  const float* wk = (const float*)d_in[3];
  const float* bk = (const float*)d_in[4];
  const float* wv = (const float*)d_in[5];
  const float* bv = (const float*)d_in[6];
  const float* wo = (const float*)d_in[7];
  const float* bo = (const float*)d_in[8];
  float* out = (float*)d_out;

  // workspace layout (bf16), all 16B-aligned
  bf16_t* p = (bf16_t*)d_ws;
  bf16_t* xb  = p; p += (size_t)M_ * E_;
  bf16_t* wqT = p; p += (size_t)E_ * P_;
  bf16_t* wkT = p; p += (size_t)E_ * P_;
  bf16_t* wvT = p; p += (size_t)E_ * P_;
  bf16_t* woT = p; p += (size_t)P_ * E_;
  bf16_t* Qb  = p; p += (size_t)M_ * P_;
  bf16_t* Kb  = p; p += (size_t)M_ * P_;
  bf16_t* Vb  = p; p += (size_t)M_ * P_;
  bf16_t* Cb  = p; p += (size_t)M_ * P_;

  // 1) converts
  cvt_bf16_kernel<<<(M_ * E_) / (256 * 4), 256, 0, stream>>>(x, xb, M_ * E_);
  dim3 tg(P_ / 32, E_ / 32);
  transpose_cvt_kernel<<<tg, 256, 0, stream>>>(wq, wqT, E_, P_);
  transpose_cvt_kernel<<<tg, 256, 0, stream>>>(wk, wkT, E_, P_);
  transpose_cvt_kernel<<<tg, 256, 0, stream>>>(wv, wvT, E_, P_);
  transpose_cvt_kernel<<<tg, 256, 0, stream>>>(wo, woT, P_, E_);

  // 2) QKV projections (bf16 out)
  dim3 gg(P_ / 128, M_ / 128);
  gemm_bf16_kernel<false><<<gg, 256, 0, stream>>>(xb, wqT, bq, Qb, M_, P_, E_);
  gemm_bf16_kernel<false><<<gg, 256, 0, stream>>>(xb, wkT, bk, Kb, M_, P_, E_);
  gemm_bf16_kernel<false><<<gg, 256, 0, stream>>>(xb, wvT, bv, Vb, M_, P_, E_);

  // 3) fused attention
  dim3 ga(N_ / 128, NH_, B_);
  attn_kernel<<<ga, 256, 0, stream>>>(Qb, Kb, Vb, Cb);

  // 4) output projection (fp32 out + bias)
  dim3 go(E_ / 128, M_ / 128);
  gemm_bf16_kernel<true><<<go, 256, 0, stream>>>(Cb, woT, bo, out, M_, E_, P_);
}
